// Model_19292993094059
// MI455X (gfx1250) — matrix operand order repaired and checked by reading, not compiled, not run on hardware
//
#include <hip/hip_runtime.h>
#include <hip/hip_bf16.h>
#include <stdint.h>

typedef __attribute__((ext_vector_type(16))) __bf16 v16bf;
typedef __attribute__((ext_vector_type(8)))  float  v8f;

#define WAVES_PER_BLOCK 4
#define THREADS (WAVES_PER_BLOCK * 32)

// padded row strides (in bf16 elements); 136*2=272B, 264*2=528B -> 4-bank rotation per row
#define PX  136   // x tile rows: 128 cols used
#define PH  264   // h tile rows: 256 cols
#define PW1 136   // W1^T rows: 128 K values
#define PW2 264   // W2^T rows: 256 K values

#define W1T_OFF 0
#define W1T_SZ  (256 * PW1)
#define W2T_OFF (W1T_OFF + W1T_SZ)
#define W2T_SZ  (64 * PW2)
#define X_OFF   (W2T_OFF + W2T_SZ)
#define X_SZ    (WAVES_PER_BLOCK * 16 * PX)
#define H_OFF   (X_OFF + X_SZ)
#define H_SZ    (WAVES_PER_BLOCK * 16 * PH)
#define US_TOTAL (H_OFF + H_SZ)   // 77312 bf16 = 154,624 B  (< 320 KB WGP LDS)

// pack two f32 -> packed bf16 dword via native conversion (v_cvt_pk_bf16_f32 class)
__device__ __forceinline__ uint32_t pk_bf16(float lo, float hi) {
    union { __bf16 h[2]; uint32_t u; } p;
    p.h[0] = (__bf16)lo;
    p.h[1] = (__bf16)hi;
    return p.u;
}

// build a 16-element bf16 WMMA operand from two 16-byte LDS chunks
__device__ __forceinline__ v16bf ld16(const __bf16* p0, const __bf16* p1) {
    union { v16bf v; uint4 q[2]; } u;
    u.q[0] = *(const uint4*)p0;
    u.q[1] = *(const uint4*)p1;
    return u.v;
}

__global__ __launch_bounds__(THREADS, 1)
void item_tower_kernel(const int* __restrict__ item_ids,
                       const int* __restrict__ tag_ids,
                       const int* __restrict__ tag_lens,
                       const float* __restrict__ item_table,
                       const float* __restrict__ item_dat_table,
                       const float* __restrict__ tag_table,
                       const float* __restrict__ W1, const float* __restrict__ b1,
                       const float* __restrict__ W2, const float* __restrict__ b2,
                       float* __restrict__ out, float* __restrict__ edat_out)
{
    __shared__ __align__(16) __bf16 smem[US_TOTAL];
    __shared__ float sb1[256];
    __shared__ float sb2[64];

    const int tid  = threadIdx.x;
    const int lane = tid & 31;
    const int wave = tid >> 5;
    const int m    = lane & 15;     // A row / B col / C-D col
    const int hi   = lane >> 4;     // lane half selects K sub-block / C-D row half

    // ---- stage weights to LDS, transposed + bf16 (only K rows 0..127 of W1 matter:
    //      tower input cols 128..191 are zeros) ----
    for (int idx = tid; idx < 128 * 256; idx += THREADS) {
        int k = idx >> 8, n = idx & 255;
        smem[W1T_OFF + n * PW1 + k] = (__bf16)W1[k * 256 + n];
    }
    for (int idx = tid; idx < 256 * 64; idx += THREADS) {
        int k = idx >> 6, n = idx & 63;
        smem[W2T_OFF + n * PW2 + k] = (__bf16)W2[k * 64 + n];
    }
    for (int i = tid; i < 256; i += THREADS) sb1[i] = b1[i];
    if (tid < 64) sb2[tid] = b2[tid];

    // ---- pooling: per-wave 16-item tile; each lane owns 2 columns (float2) ----
    const int tileIdx = blockIdx.x * WAVES_PER_BLOCK + wave;
    const int base    = tileIdx * 16;
    __bf16* xw = smem + X_OFF + wave * 16 * PX;
    const int c = lane * 2;

    for (int i = 0; i < 16; ++i) {
        const int id  = item_ids[base + i];   // uniform -> scalarized
        const int len = tag_lens[base + i];
        const float2 ei = *(const float2*)(item_table     + (size_t)id * 64 + c);
        const float2 ed = *(const float2*)(item_dat_table + (size_t)id * 64 + c);
        float ax = 0.f, ay = 0.f;
        for (int t = 0; t < len; ++t) {       // uniform trip count
            const int tg = tag_ids[(base + i) * 8 + t];
            const float2 te = *(const float2*)(tag_table + (size_t)tg * 64 + c);
            ax += te.x; ay += te.y;
        }
        const float inv = 1.0f / (float)len;
        *(uint32_t*)(xw + i * PX + c)      = pk_bf16(ei.x, ei.y);           // cols [0,64)
        *(uint32_t*)(xw + i * PX + 64 + c) = pk_bf16(ax * inv, ay * inv);   // cols [64,128)
        *(float2*)(edat_out + (size_t)(base + i) * 64 + c) = ed;  // second output
    }

    __syncthreads();   // weights visible to all waves

    // ---- GEMM1: h[16x256] = relu(x[16x128] @ W1[128x256] + b1) ----
    __bf16* hw = smem + H_OFF + wave * 16 * PH;
    v16bf ax4[4];
    {
        const __bf16* arow = xw + m * PX;
        #pragma unroll
        for (int kb = 0; kb < 4; ++kb) {
            const int k0 = kb * 32 + hi * 8;           // A: K pairs, split by lane half
            ax4[kb] = ld16(arow + k0, arow + k0 + 16);
        }
    }
    for (int j = 0; j < 16; ++j) {
        v8f acc = {};
        const __bf16* brow = smem + W1T_OFF + (j * 16 + m) * PW1 + hi * 8;
        #pragma unroll
        for (int kb = 0; kb < 4; ++kb) {
            v16bf b = ld16(brow + kb * 32, brow + kb * 32 + 16);
            acc = __builtin_amdgcn_wmma_f32_16x16x32_bf16(false, ax4[kb], false, b,
                                                          (short)0, acc, false, false);
        }
        const float bias = sb1[j * 16 + m];
        #pragma unroll
        for (int r = 0; r < 8; ++r) {
            hw[(r + hi * 8) * PH + j * 16 + m] = (__bf16)fmaxf(acc[r] + bias, 0.0f);
        }
    }

    // ---- GEMM2: out[16x64] = h[16x256] @ W2[256x64] + b2 ----
    v16bf ah[8];
    {
        const __bf16* arow = hw + m * PH;
        #pragma unroll
        for (int kb = 0; kb < 8; ++kb) {
            const int k0 = kb * 32 + hi * 8;
            ah[kb] = ld16(arow + k0, arow + k0 + 16);
        }
    }
    for (int j = 0; j < 4; ++j) {
        v8f acc = {};
        const __bf16* brow = smem + W2T_OFF + (j * 16 + m) * PW2 + hi * 8;
        #pragma unroll
        for (int kb = 0; kb < 8; ++kb) {
            v16bf b = ld16(brow + kb * 32, brow + kb * 32 + 16);
            acc = __builtin_amdgcn_wmma_f32_16x16x32_bf16(false, ah[kb], false, b,
                                                          (short)0, acc, false, false);
        }
        const float bias = sb2[j * 16 + m];
        #pragma unroll
        for (int r = 0; r < 8; ++r) {
            out[(size_t)(base + r + hi * 8) * 64 + j * 16 + m] = acc[r] + bias;
        }
    }
}

extern "C" void kernel_launch(void* const* d_in, const int* in_sizes, int n_in,
                              void* d_out, int out_size, void* d_ws, size_t ws_size,
                              hipStream_t stream) {
    const int*   item_ids   = (const int*)d_in[0];
    const int*   tag_ids    = (const int*)d_in[1];
    const int*   tag_lens   = (const int*)d_in[2];
    const float* item_table = (const float*)d_in[3];
    const float* item_dat   = (const float*)d_in[4];
    const float* tag_table  = (const float*)d_in[5];
    const float* W1         = (const float*)d_in[6];
    const float* b1         = (const float*)d_in[7];
    const float* W2         = (const float*)d_in[8];
    const float* b2         = (const float*)d_in[9];

    const int N = in_sizes[0];               // 16384
    float* out  = (float*)d_out;             // [N,64]
    float* edat = out + (size_t)N * 64;      // [N,64]

    const int blocks = N / (WAVES_PER_BLOCK * 16);   // 256
    hipLaunchKernelGGL(item_tower_kernel, dim3(blocks), dim3(THREADS), 0, stream,
                       item_ids, tag_ids, tag_lens, item_table, item_dat, tag_table,
                       W1, b1, W2, b2, out, edat);
}
